// MMDecoderLSTM_5549097746440
// MI455X (gfx1250) — compile-verified, matrix operand order checked by reading, not run on hardware
//
#include <hip/hip_runtime.h>
#include <cmath>

// ---------------- problem constants ----------------
#define SEQ_LEN   30
#define H_DIM     64
#define EMB       16
#define NSAMP     3
#define TRAJ_PTS  6
#define BATCHN    131072

// ---------------- tiling constants ----------------
#define KDIM      96      // [ x(16) | h(64) | pad(16) ]  -> 3 WMMA K=32 chunks
#define NROWS     272     // 256 gate rows | 6 h2p | 3 conf | 7 pad  -> 17 N-tiles
#define WB_HALFS  (NROWS * KDIM)            // 26112 f16
#define WB_BYTES  (WB_HALFS * 2)            // 52224
#define AST_BYTES_PW (16 * KDIM * 2)        // 3072  per-wave A staging
#define PRD_FLOATS_PW (16 * 180)            // per-wave pred accumulation
#define PRD_BYTES_PW (PRD_FLOATS_PW * 4)    // 11520
#define WAVES_PER_BLOCK 8
#define SMEM_BYTES (WB_BYTES + WAVES_PER_BLOCK * (AST_BYTES_PW + PRD_BYTES_PW)) // 168960

// gfx1250 async global->LDS path (guarded: falls back to plain loads)
#if defined(__has_builtin)
#  if __has_builtin(__builtin_amdgcn_global_load_async_to_lds_b128)
#    define HAVE_ASYNC_LDS 1
#  endif
#endif

typedef __attribute__((ext_vector_type(16))) _Float16 v16h;
typedef __attribute__((ext_vector_type(8)))  _Float16 v8h;
typedef __attribute__((ext_vector_type(8)))  float    v8f;

#if HAVE_ASYNC_LDS
// Exact parameter types of __builtin_amdgcn_global_load_async_to_lds_b128:
//   (v4i addrspace(1)*, v4i addrspace(3)*, imm int offset, imm int cpol)
typedef int v4i_vs __attribute__((vector_size(16)));
typedef __attribute__((address_space(1))) v4i_vs* g_v4i_p;
typedef __attribute__((address_space(3))) v4i_vs* l_v4i_p;
#endif

union Frag16 { v16h v; v8h h[2]; };

// Wave-local LDS ordering: DS ops are in-order per wave on CDNA5; the
// s_wait_dscnt + memory clobber stops the compiler reordering across it.
__device__ __forceinline__ void lds_fence() {
  asm volatile("s_wait_dscnt 0x0" ::: "memory");
}

// Branch-free activations on the HW transcendental unit (v_exp_f32 is 2^x).
// Saturate correctly: exp2->inf gives sigm->0/1, tanh->+/-1.
#define LOG2E  1.4426950408889634f
__device__ __forceinline__ float sigm(float x) {
  return __builtin_amdgcn_rcpf(1.0f + __builtin_amdgcn_exp2f(-LOG2E * x));
}
__device__ __forceinline__ float fast_tanh(float x) {
  return 1.0f - 2.0f * __builtin_amdgcn_rcpf(1.0f + __builtin_amdgcn_exp2f((2.0f * LOG2E) * x));
}
__device__ __forceinline__ float lrelu(float x) { return x > 0.0f ? x : 0.01f * x; }

__device__ __forceinline__ v8f vzero8() {
  v8f z;
#pragma unroll
  for (int i = 0; i < 8; ++i) z[i] = 0.0f;
  return z;
}

__global__ void __launch_bounds__(256)
mmdec_lstm_wmma(const float* __restrict__ traj_rel,
                const float* __restrict__ h0,
                const float* __restrict__ c0,
                const float* __restrict__ W_ih,
                const float* __restrict__ W_hh,
                const float* __restrict__ b_ih,
                const float* __restrict__ b_hh,
                const float* __restrict__ W_se,
                const float* __restrict__ b_se,
                const float* __restrict__ W_h2p,
                const float* __restrict__ b_h2p,
                const float* __restrict__ W_conf,
                const float* __restrict__ b_conf,
                float* __restrict__ out)
{
  extern __shared__ char smem[];
  _Float16* WB = (_Float16*)smem;  // [NROWS][KDIM] f16 B-matrix image

  const int tid  = threadIdx.x;
  const int lane = tid & 31;
  const int wv   = tid >> 5;        // wave in block, 0..7
  const int lh   = lane & 15;       // lane%16
  const int sg   = lane >> 4;       // lane half (0/1)

  _Float16* AST = (_Float16*)(smem + WB_BYTES) + wv * (16 * KDIM);
  float*    PRD = (float*)(smem + WB_BYTES + WAVES_PER_BLOCK * AST_BYTES_PW)
                  + wv * PRD_FLOATS_PW;

  const size_t b0 = ((size_t)blockIdx.x * WAVES_PER_BLOCK + wv) * 16;

  // ---- stage f32 LSTM weights (80KB) into LDS ----
  // The PRD region is dead until step 0, so reuse it as an f32 staging area.
  const float* wih_src;
  const float* whh_src;
#if HAVE_ASYNC_LDS
  {
    float* STG = (float*)(smem + WB_BYTES + WAVES_PER_BLOCK * AST_BYTES_PW);
    // 1024 float4 for W_ih (256x16), 4096 float4 for W_hh (256x64)
    for (int idx = tid; idx < 5120; idx += 256) {
      const float* src = (idx < 1024) ? (W_ih + idx * 4) : (W_hh + (idx - 1024) * 4);
      __builtin_amdgcn_global_load_async_to_lds_b128(
          (g_v4i_p)(uintptr_t)src,
          (l_v4i_p)(unsigned int)(uintptr_t)(STG + idx * 4),
          0, 0);
    }
    asm volatile("s_wait_asynccnt 0x0" ::: "memory");
    __syncthreads();   // all async tiles visible block-wide
    wih_src = STG;
    whh_src = STG + 4096;
  }
#else
  __builtin_prefetch(W_hh + tid, 0, 0);
  __builtin_prefetch(W_ih + tid, 0, 0);
  wih_src = W_ih;
  whh_src = W_hh;
#endif

  // ---- build f16 weight image in LDS (block cooperative) ----
  for (int idx = tid; idx < WB_HALFS; idx += 256) {
    const int n = idx / KDIM;
    const int k = idx % KDIM;
    float w = 0.0f;
    if (n < 256) {
      if (k < 16)       w = wih_src[n * EMB + k];
      else if (k < 80)  w = whh_src[n * H_DIM + (k - 16)];
    } else if (n < 262) {
      if (k >= 16 && k < 80) w = W_h2p[(n - 256) * H_DIM + (k - 16)];
    } else if (n < 265) {
      if (k >= 16 && k < 80) w = W_conf[(n - 262) * H_DIM + (k - 16)];
    }
    WB[idx] = (_Float16)w;
  }
  __syncthreads();

  // ---- per-lane persistent small weights ----
  float wse[6];
#pragma unroll
  for (int n = 0; n < 6; ++n) wse[n] = W_se[lh * TRAJ_PTS + n];
  const float bj = b_se[lh];

  float biasT[17];
#pragma unroll
  for (int t = 0; t < 16; ++t) biasT[t] = b_ih[lh + 16 * t] + b_hh[lh + 16 * t];
  biasT[16] = (lh < 6) ? b_h2p[lh] : ((lh < 9) ? b_conf[lh - 6] : 0.0f);

  // ---- per-wave A-staging init ----
  // pad columns 80..95 = 0 (their weights are also 0 -> doubly safe)
  for (int idx = lane; idx < 256; idx += 32) {
    const int m = idx >> 4, k = idx & 15;
    AST[m * KDIM + 80 + k] = (_Float16)0.0f;
  }
  // h0 -> columns 16..79
  for (int idx = lane; idx < 16 * 64; idx += 32) {
    const int m = idx >> 6, k = idx & 63;
    AST[m * KDIM + 16 + k] = (_Float16)h0[(b0 + m) * H_DIM + k];
  }
  // x0 = leaky_relu(tile(traj_rel,3) @ W_se^T + b_se): lane owns column j=lh
#pragma unroll
  for (int mm = 0; mm < 8; ++mm) {
    const int m = sg * 8 + mm;
    const float tx = traj_rel[(b0 + m) * 2 + 0];
    const float ty = traj_rel[(b0 + m) * 2 + 1];
    float acc = bj + tx * (wse[0] + wse[2] + wse[4]) + ty * (wse[1] + wse[3] + wse[5]);
    AST[m * KDIM + lh] = (_Float16)lrelu(acc);
  }

  // ---- c state in C-fragment layout: cf[t][r] = c[m = r+8*sg][n = lh+16t] ----
  v8f cf[4];
#pragma unroll
  for (int t = 0; t < 4; ++t)
#pragma unroll
    for (int r = 0; r < 8; ++r)
      cf[t][r] = c0[(b0 + r + 8 * sg) * H_DIM + lh + 16 * t];

  lds_fence();

  // =======================  recurrent loop  =======================
  for (int s = 0; s < SEQ_LEN; ++s) {
    // A fragments: M = lh, K-groups (sg*8..+7) and (16+sg*8..+7) per chunk
    Frag16 a0, a1, a2;
    {
      const _Float16* ar = AST + lh * KDIM;
      a0.h[0] = *(const v8h*)(ar +  0 + sg * 8);
      a0.h[1] = *(const v8h*)(ar + 16 + sg * 8);
      a1.h[0] = *(const v8h*)(ar + 32 + sg * 8);
      a1.h[1] = *(const v8h*)(ar + 48 + sg * 8);
      a2.h[0] = *(const v8h*)(ar + 64 + sg * 8);
      a2.h[1] = *(const v8h*)(ar + 80 + sg * 8);
    }

    // gates GEMM per hidden group: tiles (i,f,g,o) = (hg, 4+hg, 8+hg, 12+hg)
#pragma unroll
    for (int hg = 0; hg < 4; ++hg) {
      v8f g4[4];
#pragma unroll
      for (int q = 0; q < 4; ++q) {
        const int t = q * 4 + hg;
        const _Float16* wrow = WB + (t * 16 + lh) * KDIM;
        v8f acc = vzero8();
        Frag16 bf;
        bf.h[0] = *(const v8h*)(wrow +  0 + sg * 8);
        bf.h[1] = *(const v8h*)(wrow + 16 + sg * 8);
        acc = __builtin_amdgcn_wmma_f32_16x16x32_f16(false, a0.v, false, bf.v,
                                                     (short)0, acc, false, false);
        bf.h[0] = *(const v8h*)(wrow + 32 + sg * 8);
        bf.h[1] = *(const v8h*)(wrow + 48 + sg * 8);
        acc = __builtin_amdgcn_wmma_f32_16x16x32_f16(false, a1.v, false, bf.v,
                                                     (short)0, acc, false, false);
        bf.h[0] = *(const v8h*)(wrow + 64 + sg * 8);
        bf.h[1] = *(const v8h*)(wrow + 80 + sg * 8);
        acc = __builtin_amdgcn_wmma_f32_16x16x32_f16(false, a2.v, false, bf.v,
                                                     (short)0, acc, false, false);
        const float bt = biasT[t];
#pragma unroll
        for (int r = 0; r < 8; ++r) acc[r] += bt;
        g4[q] = acc;
      }
      // lane-local LSTM pointwise update; write new h (f16) to A staging
#pragma unroll
      for (int r = 0; r < 8; ++r) {
        const float ii = sigm(g4[0][r]);
        const float ff = sigm(g4[1][r]);
        const float gg = fast_tanh(g4[2][r]);
        const float oo = sigm(g4[3][r]);
        const float c  = ff * cf[hg][r] + ii * gg;
        cf[hg][r] = c;
        const float h = oo * fast_tanh(c);
        AST[(r + 8 * sg) * KDIM + 16 + hg * 16 + lh] = (_Float16)h;
      }
    }

    lds_fence();

    // hidden2pos + conf via WMMA tile 16 (x-column weights are zero)
    {
      const _Float16* ar = AST + lh * KDIM;
      a0.h[0] = *(const v8h*)(ar +  0 + sg * 8);
      a0.h[1] = *(const v8h*)(ar + 16 + sg * 8);
      a1.h[0] = *(const v8h*)(ar + 32 + sg * 8);
      a1.h[1] = *(const v8h*)(ar + 48 + sg * 8);
      a2.h[0] = *(const v8h*)(ar + 64 + sg * 8);
      a2.h[1] = *(const v8h*)(ar + 80 + sg * 8);
    }
    v8f rc = vzero8();
    {
      const _Float16* wrow = WB + (256 + lh) * KDIM;
      Frag16 bf;
      bf.h[0] = *(const v8h*)(wrow +  0 + sg * 8);
      bf.h[1] = *(const v8h*)(wrow + 16 + sg * 8);
      rc = __builtin_amdgcn_wmma_f32_16x16x32_f16(false, a0.v, false, bf.v,
                                                  (short)0, rc, false, false);
      bf.h[0] = *(const v8h*)(wrow + 32 + sg * 8);
      bf.h[1] = *(const v8h*)(wrow + 48 + sg * 8);
      rc = __builtin_amdgcn_wmma_f32_16x16x32_f16(false, a1.v, false, bf.v,
                                                  (short)0, rc, false, false);
      bf.h[0] = *(const v8h*)(wrow + 64 + sg * 8);
      bf.h[1] = *(const v8h*)(wrow + 80 + sg * 8);
      rc = __builtin_amdgcn_wmma_f32_16x16x32_f16(false, a2.v, false, bf.v,
                                                  (short)0, rc, false, false);
    }
    {
      const float bt = biasT[16];
#pragma unroll
      for (int r = 0; r < 8; ++r) rc[r] += bt;
    }

    // rel -> pred LDS buffer in final (3,30,2) per-row layout
    if (lh < 6) {
      const int i = lh >> 1, d = lh & 1;
      const int off = i * 60 + s * 2 + d;
#pragma unroll
      for (int r = 0; r < 8; ++r) PRD[(r + 8 * sg) * 180 + off] = rc[r];
    } else if (s == SEQ_LEN - 1 && lh < 9) {
      // conf logits: stash in (now dead) A staging, reinterpreted as f32
      float* CONFB = (float*)AST;
#pragma unroll
      for (int r = 0; r < 8; ++r) CONFB[(r + 8 * sg) * 3 + (lh - 6)] = rc[r];
    }
    lds_fence();

    // next x = leaky_relu(rel @ W_se^T + b_se): lane owns column j=lh
    if (s < SEQ_LEN - 1) {
#pragma unroll
      for (int mm = 0; mm < 8; ++mm) {
        const int m = sg * 8 + mm;
        const float* pr = PRD + m * 180;
        const float2 p0 = *(const float2*)(pr + 0 * 60 + s * 2);
        const float2 p1 = *(const float2*)(pr + 1 * 60 + s * 2);
        const float2 p2 = *(const float2*)(pr + 2 * 60 + s * 2);
        float acc = bj + p0.x * wse[0] + p0.y * wse[1]
                       + p1.x * wse[2] + p1.y * wse[3]
                       + p2.x * wse[4] + p2.y * wse[5];
        AST[m * KDIM + lh] = (_Float16)lrelu(acc);
      }
    }
    lds_fence();
  }

  // ---- conf softmax (lanes 0..15, one batch row each) ----
  if (sg == 0) {
    const float* CONFB = (const float*)AST;
    const float l0 = CONFB[lh * 3 + 0];
    const float l1 = CONFB[lh * 3 + 1];
    const float l2 = CONFB[lh * 3 + 2];
    const float mx = fmaxf(l0, fmaxf(l1, l2));
    const float e0 = __builtin_amdgcn_exp2f(LOG2E * (l0 - mx));
    const float e1 = __builtin_amdgcn_exp2f(LOG2E * (l1 - mx));
    const float e2 = __builtin_amdgcn_exp2f(LOG2E * (l2 - mx));
    const float inv = __builtin_amdgcn_rcpf(e0 + e1 + e2);
    float* cout = out + (size_t)BATCHN * 180 + (b0 + lh) * 3;
    cout[0] = e0 * inv; cout[1] = e1 * inv; cout[2] = e2 * inv;
  }

  // ---- coalesced pred flush: 16 rows x 180 f32 as float4 ----
  for (int idx = lane; idx < 16 * 45; idx += 32) {
    const int m = idx / 45, c4 = idx % 45;
    *(float4*)(out + (b0 + m) * 180 + c4 * 4) =
        *(const float4*)(PRD + m * 180 + c4 * 4);
  }
}

extern "C" void kernel_launch(void* const* d_in, const int* in_sizes, int n_in,
                              void* d_out, int out_size, void* d_ws, size_t ws_size,
                              hipStream_t stream) {
  (void)in_sizes; (void)n_in; (void)out_size; (void)d_ws; (void)ws_size;
  const float* traj_rel = (const float*)d_in[1];
  const float* h0     = (const float*)d_in[2];
  const float* c0     = (const float*)d_in[3];
  const float* W_ih   = (const float*)d_in[4];
  const float* W_hh   = (const float*)d_in[5];
  const float* b_ih   = (const float*)d_in[6];
  const float* b_hh   = (const float*)d_in[7];
  const float* W_se   = (const float*)d_in[8];
  const float* b_se   = (const float*)d_in[9];
  const float* W_h2p  = (const float*)d_in[10];
  const float* b_h2p  = (const float*)d_in[11];
  const float* W_conf = (const float*)d_in[12];
  const float* b_conf = (const float*)d_in[13];
  float* out = (float*)d_out;

  // CDNA5 allows up to 320KB LDS per workgroup; we request 169KB dynamically.
  (void)hipFuncSetAttribute((const void*)mmdec_lstm_wmma,
                            hipFuncAttributeMaxDynamicSharedMemorySize,
                            SMEM_BYTES);

  const int rows_per_block = 16 * WAVES_PER_BLOCK;       // 128
  dim3 grid(BATCHN / rows_per_block);                    // 1024
  dim3 block(256);                                       // 8 waves (wave32)
  mmdec_lstm_wmma<<<grid, block, SMEM_BYTES, stream>>>(
      traj_rel, h0, c0, W_ih, W_hh, b_ih, b_hh,
      W_se, b_se, W_h2p, b_h2p, W_conf, b_conf, out);
}
